// SelfAttention_11484742549780
// MI455X (gfx1250) — matrix operand order repaired and checked by reading, not compiled, not run on hardware
//
#include <hip/hip_runtime.h>
#include <cstdint>
#include <cstddef>

// ---------------- problem constants ----------------
#define S_LEN   2048
#define BATCH   2
#define HDIM    4096
#define NHEADS  32
#define NGROUPS 2
#define HEADD   128
#define MROWS   (S_LEN * BATCH)                   // 4096 flattened (s,b) rows
#define NQKV    (NHEADS*HEADD + 2*NGROUPS*HEADD)  // 4608

// ---------------- vector types (all trivially-copyable) ----------------
typedef __attribute__((ext_vector_type(16))) __bf16        v16bf;
typedef __attribute__((ext_vector_type(8)))  float         v8f;
typedef __attribute__((ext_vector_type(4)))  unsigned int  vu4;
typedef __attribute__((ext_vector_type(2)))  unsigned int  vu2;
typedef __attribute__((ext_vector_type(4)))  float         vf4;

union Frag  { v16bf v; vu4 u[2]; };
union Pack4 { __bf16 h[4]; vu2 u; };
union Pack8 { __bf16 h[8]; vu4 u; };

__device__ __forceinline__ v8f wmma_bf16(const Frag& a, const Frag& b, v8f c) {
  // D(f32 16x16) = A(bf16 16x32) * B(bf16 32x16) + C
  return __builtin_amdgcn_wmma_f32_16x16x32_bf16(false, a.v, false, b.v,
                                                 (short)0, c, false, false);
}

// ---------------- async global->LDS (CDNA5 ASYNCcnt path) ----------------
#define AS1 __attribute__((address_space(1)))
#define AS3 __attribute__((address_space(3)))

// the builtin's pointer params are GCC-style int4 vectors (per clang diag)
typedef int v4i_async __attribute__((vector_size(16)));

#if defined(__has_builtin)
#if __has_builtin(__builtin_amdgcn_global_load_async_to_lds_b128) && \
    __has_builtin(__builtin_amdgcn_s_wait_asynccnt)
#define HAVE_ASYNC_LDS 1
#endif
#endif
#ifndef HAVE_ASYNC_LDS
#define HAVE_ASYNC_LDS 0
#endif

__device__ __forceinline__ void async_copy_b128(const void* g, void* l) {
#if HAVE_ASYNC_LDS
  // AS3 ptr == low 32 bits of the generic LDS address (ISA 10.2 aperture rule);
  // AS1 ptr is numerically identical to the generic global address.
  __builtin_amdgcn_global_load_async_to_lds_b128(
      (AS1 v4i_async*)(uintptr_t)g,
      (AS3 v4i_async*)(uint32_t)(uintptr_t)l, 0, 0);
#else
  *(vu4*)l = *(const vu4*)g;
#endif
}

__device__ __forceinline__ void async_wait() {
#if HAVE_ASYNC_LDS
  __builtin_amdgcn_s_wait_asynccnt(0);
#endif
}

// =====================================================================
// Kernel 1: QKV projection.  mixed = hs @ Wqkv^T + b, scattered as bf16
// into Q[B][NH][S][HD], K[B][NG][S][HD], V[B][NG][S][HD].
// Block = 256 thr (8 waves), tile 128(M) x 128(N), K-tile 64.
// =====================================================================
__global__ __launch_bounds__(256) void qkv_gemm_kernel(
    const float* __restrict__ A,      // (4096, 4096) row-major (m = s*B+b)
    const float* __restrict__ W,      // (4608, 4096) row-major
    const float* __restrict__ bias,   // (4608,)
    __bf16* __restrict__ Qb, __bf16* __restrict__ Kb, __bf16* __restrict__ Vb)
{
  __shared__ __bf16 As[128][72];
  __shared__ __bf16 Bs[128][72];

  const int tid  = threadIdx.x;
  const int lane = tid & 31, wid = tid >> 5;
  const int wm   = wid & 1,  wn  = wid >> 1;     // 2 x 4 wave grid
  const int ml   = lane & 15, half = lane >> 4;
  const int bm   = blockIdx.y, bn = blockIdx.x;

  v8f acc[8];
  #pragma unroll
  for (int i = 0; i < 8; ++i)
    #pragma unroll
    for (int r = 0; r < 8; ++r) acc[i][r] = 0.f;

  const int srow = tid >> 4;          // 16 threads per row, 4 fp32 each
  const int scol = (tid & 15) * 4;

  for (int k0 = 0; k0 < HDIM; k0 += 64) {
    __syncthreads();
    if (k0 + 64 < HDIM) {             // prefetch next K-tile (global_prefetch)
      __builtin_prefetch(A + (size_t)(bm*128 + srow)*HDIM + k0 + 64 + scol, 0, 0);
      __builtin_prefetch(W + (size_t)(bn*128 + srow)*HDIM + k0 + 64 + scol, 0, 0);
    }
    #pragma unroll
    for (int p = 0; p < 8; ++p) {
      const int r = srow + p * 16;
      vf4 fa = *(const vf4*)(A + (size_t)(bm*128 + r)*HDIM + k0 + scol);
      vf4 fb = *(const vf4*)(W + (size_t)(bn*128 + r)*HDIM + k0 + scol);
      Pack4 pa, pb;
      pa.h[0]=(__bf16)fa.x; pa.h[1]=(__bf16)fa.y; pa.h[2]=(__bf16)fa.z; pa.h[3]=(__bf16)fa.w;
      pb.h[0]=(__bf16)fb.x; pb.h[1]=(__bf16)fb.y; pb.h[2]=(__bf16)fb.z; pb.h[3]=(__bf16)fb.w;
      *(vu2*)&As[r][scol] = pa.u;
      *(vu2*)&Bs[r][scol] = pb.u;
    }
    __syncthreads();

    #pragma unroll
    for (int kk = 0; kk < 64; kk += 32) {
      Frag af[4], bf2[2];
      #pragma unroll
      for (int mt = 0; mt < 4; ++mt) {
        const __bf16* p = &As[wm*64 + mt*16 + ml][kk + half*8];
        af[mt].u[0] = *(const vu4*)p;
        af[mt].u[1] = *(const vu4*)(p + 16);
      }
      #pragma unroll
      for (int nt = 0; nt < 2; ++nt) {
        const __bf16* p = &Bs[wn*32 + nt*16 + ml][kk + half*8];
        bf2[nt].u[0] = *(const vu4*)p;
        bf2[nt].u[1] = *(const vu4*)(p + 16);
      }
      #pragma unroll
      for (int mt = 0; mt < 4; ++mt)
        #pragma unroll
        for (int nt = 0; nt < 2; ++nt)
          acc[mt*2+nt] = wmma_bf16(af[mt], bf2[nt], acc[mt*2+nt]);
    }
  }

  // epilogue: block-uniform head routing (128 cols == 1 head since HD==128)
  int kind, hx;
  if      (bn < NHEADS)     { kind = 0; hx = bn; }
  else if (bn < NHEADS + 2) { kind = 1; hx = bn - NHEADS; }
  else                      { kind = 2; hx = bn - NHEADS - 2; }
  __bf16* base = (kind == 0) ? Qb : ((kind == 1) ? Kb : Vb);
  const int NHx = (kind == 0) ? NHEADS : NGROUPS;

  float bv[2];
  #pragma unroll
  for (int nt = 0; nt < 2; ++nt) bv[nt] = bias[bn*128 + wn*32 + nt*16 + ml];

  #pragma unroll
  for (int mt = 0; mt < 4; ++mt)
    #pragma unroll
    for (int nt = 0; nt < 2; ++nt)
      #pragma unroll
      for (int r = 0; r < 8; ++r) {
        const int mrow = bm*128 + wm*64 + mt*16 + r + half*8;
        const int s = mrow >> 1, b = mrow & 1;        // m = s*B + b, B==2
        const int d = wn*32 + nt*16 + ml;
        const float v = acc[mt*2+nt][r] + bv[nt];
        base[((size_t)(b*NHx + hx)*S_LEN + s)*HEADD + d] = (__bf16)v;
      }
}

// =====================================================================
// Kernel 2: RoPE in-place on bf16 Q and K (first 64 dims, pairs (2i,2i+1))
// =====================================================================
__global__ __launch_bounds__(256) void rope_kernel(
    __bf16* __restrict__ Qb, __bf16* __restrict__ Kb,
    const float* __restrict__ rope)   // (S,1,32,2) flat: s*64 + i*2 + {c,s}
{
  const int NQP = BATCH*NHEADS*S_LEN*32;   // 4,194,304 pairs
  const int NKP = BATCH*NGROUPS*S_LEN*32;  //   262,144 pairs
  int idx = blockIdx.x * 256 + threadIdx.x;
  __bf16* base; int hb, rem;
  if (idx < NQP)            { base = Qb; hb = idx >> 16;  rem = idx & 65535; }
  else if (idx < NQP + NKP) { int j = idx - NQP; base = Kb; hb = j >> 16; rem = j & 65535; }
  else return;
  const int s = rem >> 5, i = rem & 31;
  const size_t a = ((size_t)hb*S_LEN + s)*HEADD + i*2;
  const float c  = rope[s*64 + i*2];
  const float sn = rope[s*64 + i*2 + 1];
  const float x0 = (float)base[a], x1 = (float)base[a+1];
  base[a]   = (__bf16)(x0*c - x1*sn);
  base[a+1] = (__bf16)(x1*c + x0*sn);
}

// =====================================================================
// Kernel 3: causal flash attention. Block = 128 thr (4 waves), 64 queries
// per block (16 q-rows per wave), 64-key tiles, HD=128.
// K tile staged via async global->LDS (ASYNCcnt); V staged transposed.
// grid = (S/64, B*NH)
// =====================================================================
__global__ __launch_bounds__(128) void attn_kernel(
    const __bf16* __restrict__ Qb, const __bf16* __restrict__ Kb,
    const __bf16* __restrict__ Vb, __bf16* __restrict__ Cx)  // ctx (4096,4096) bf16
{
  __shared__ __bf16 Ks[64][136];   // K tile: [t][d]
  __shared__ __bf16 VT[128][72];   // V^T tile: [d][t]
  __shared__ __bf16 Pl[64][72];    // probs tile: [q_local][t_local]

  const int tid  = threadIdx.x;
  const int lane = tid & 31, w = tid >> 5;
  const int ml   = lane & 15, half = lane >> 4;
  const int q0   = blockIdx.x * 64;
  const int bh   = blockIdx.y;                 // b*32 + h
  const int b    = bh >> 5, h = bh & 31, g = h >> 4;

  const __bf16* Qg = Qb + (size_t)bh * S_LEN * HEADD;
  const __bf16* Kg = Kb + (size_t)(b*NGROUPS + g) * S_LEN * HEADD;
  const __bf16* Vg = Vb + (size_t)(b*NGROUPS + g) * S_LEN * HEADD;

  // Q A-fragments held in registers across the whole kernel
  Frag qf[4];
  {
    const __bf16* Qrow = Qg + (size_t)(q0 + w*16 + ml) * HEADD;
    #pragma unroll
    for (int c = 0; c < 4; ++c) {
      qf[c].u[0] = *(const vu4*)(Qrow + c*32 + half*8);
      qf[c].u[1] = *(const vu4*)(Qrow + c*32 + half*8 + 16);
    }
  }

  v8f cacc[8];
  #pragma unroll
  for (int i = 0; i < 8; ++i)
    #pragma unroll
    for (int r = 0; r < 8; ++r) cacc[i][r] = 0.f;
  float mst[8], lst[8];
  #pragma unroll
  for (int r = 0; r < 8; ++r) { mst[r] = -1e30f; lst[r] = 0.f; }

  const float SCALE = 0.08838834764831845f;   // 1/sqrt(128)
  const int vtr = tid >> 4;                   // staging: 16 thr per key row
  const int vdc = (tid & 15) * 8;             // 8 bf16 (16B) per thread

  for (int t0 = 0; t0 <= q0; t0 += 64) {
    __syncthreads();                          // protect Ks/VT vs previous reads

    if (t0 + 64 <= q0) {                      // prefetch next key tile
      __builtin_prefetch(Kg + (size_t)(t0 + 64 + vtr)*HEADD + vdc, 0, 0);
      __builtin_prefetch(Vg + (size_t)(t0 + 64 + vtr)*HEADD + vdc, 0, 0);
    }

    // async-stage K tile (64 x 128 bf16) straight into LDS
    #pragma unroll
    for (int p = 0; p < 8; ++p) {
      const int t = vtr + p*8;
      async_copy_b128(Kg + (size_t)(t0 + t)*HEADD + vdc, &Ks[t][vdc]);
    }
    // stage V^T (64 keys x 128 d -> VT[d][t])
    #pragma unroll
    for (int p = 0; p < 8; ++p) {
      const int t = vtr + p*8;
      Pack8 cv; cv.u = *(const vu4*)(Vg + (size_t)(t0 + t)*HEADD + vdc);
      #pragma unroll
      for (int j = 0; j < 8; ++j) VT[vdc + j][t] = cv.h[j];
    }
    async_wait();                             // drain ASYNCcnt before barrier
    __syncthreads();                          // Ks + VT visible to all waves

    // scores: 16 q-rows x 64 keys, reduce over HD=128 (K B-frags from LDS)
    v8f sacc[4];
    #pragma unroll
    for (int i = 0; i < 4; ++i)
      #pragma unroll
      for (int r = 0; r < 8; ++r) sacc[i][r] = 0.f;

    #pragma unroll
    for (int c = 0; c < 4; ++c) {
      #pragma unroll
      for (int nt = 0; nt < 4; ++nt) {
        Frag kf;
        const __bf16* Krow = &Ks[nt*16 + ml][c*32 + half*8];
        kf.u[0] = *(const vu4*)Krow;
        kf.u[1] = *(const vu4*)(Krow + 16);
        sacc[nt] = wmma_bf16(qf[c], kf, sacc[nt]);
      }
    }

    // online softmax (rows live in 16-lane halves: m = r + half*8)
    const bool diag = (t0 == q0);
    const int qlb = w*16 + half*8;            // + r = local query row (abs - q0)
    #pragma unroll
    for (int r = 0; r < 8; ++r) {
      float mx = -1e30f;
      #pragma unroll
      for (int nt = 0; nt < 4; ++nt) {
        float sv = sacc[nt][r] * SCALE;
        if (diag && (nt*16 + ml) > (qlb + r)) sv = -1e30f;
        mx = fmaxf(mx, sv);
      }
      #pragma unroll
      for (int off = 1; off <= 8; off <<= 1)
        mx = fmaxf(mx, __shfl_xor(mx, off, 32));
      const float mn   = fmaxf(mst[r], mx);
      const float corr = __expf(mst[r] - mn);
      mst[r] = mn;
      float rsum = 0.f;
      #pragma unroll
      for (int nt = 0; nt < 4; ++nt) {
        float sv = sacc[nt][r] * SCALE;
        if (diag && (nt*16 + ml) > (qlb + r)) sv = -1e30f;
        const float pe = __expf(sv - mn);
        rsum += pe;
        Pl[w*16 + half*8 + r][nt*16 + ml] = (__bf16)pe;
      }
      #pragma unroll
      for (int off = 1; off <= 8; off <<= 1)
        rsum += __shfl_xor(rsum, off, 32);
      lst[r] = lst[r] * corr + rsum;
      #pragma unroll
      for (int dt = 0; dt < 8; ++dt) cacc[dt][r] *= corr;
    }

    // PV: ctx(16 x 128) += P(16 x 64) * V(64 x 128); P rows are wave-private
    Frag pa[2];
    {
      const __bf16* Prow = &Pl[w*16 + ml][0];
      #pragma unroll
      for (int c = 0; c < 2; ++c) {
        pa[c].u[0] = *(const vu4*)(Prow + c*32 + half*8);
        pa[c].u[1] = *(const vu4*)(Prow + c*32 + half*8 + 16);
      }
    }
    #pragma unroll
    for (int dt = 0; dt < 8; ++dt) {
      #pragma unroll
      for (int c = 0; c < 2; ++c) {
        Frag vf;
        const __bf16* Vrow = &VT[dt*16 + ml][c*32 + half*8];
        vf.u[0] = *(const vu4*)Vrow;
        vf.u[1] = *(const vu4*)(Vrow + 16);
        cacc[dt] = wmma_bf16(pa[c], vf, cacc[dt]);
      }
    }
  }

  // normalize + store ctx row-major (m = s*2+b, col = h*128+d) as bf16
  #pragma unroll
  for (int r = 0; r < 8; ++r) {
    const float inv = 1.f / lst[r];
    const int   qa  = q0 + w*16 + half*8 + r;
    const size_t row = (size_t)qa * BATCH + b;
    #pragma unroll
    for (int dt = 0; dt < 8; ++dt)
      Cx[row*(size_t)(NHEADS*HEADD) + h*HEADD + dt*16 + ml] =
          (__bf16)(cacc[dt][r] * inv);
  }
}

// =====================================================================
// Kernel 4: out = ctx(bf16) @ W_dense^T, fp32 output.  A tile staged via
// async global->LDS; W converted fp32->bf16 during staging.
// =====================================================================
__global__ __launch_bounds__(256) void dense_gemm_kernel(
    const __bf16* __restrict__ A,    // (4096, 4096) bf16
    const float*  __restrict__ W,    // (4096, 4096) fp32
    float* __restrict__ out)         // (4096, 4096) fp32  == (S,B,H)
{
  __shared__ __bf16 As[128][72];
  __shared__ __bf16 Bs[128][72];

  const int tid  = threadIdx.x;
  const int lane = tid & 31, wid = tid >> 5;
  const int wm   = wid & 1,  wn  = wid >> 1;
  const int ml   = lane & 15, half = lane >> 4;
  const int bm   = blockIdx.y, bn = blockIdx.x;

  v8f acc[8];
  #pragma unroll
  for (int i = 0; i < 8; ++i)
    #pragma unroll
    for (int r = 0; r < 8; ++r) acc[i][r] = 0.f;

  const int arow = tid >> 3;          // bf16 async copy: 8 thr/row, 8 elems
  const int acol = (tid & 7) * 8;
  const int srow = tid >> 4;          // fp32->bf16: 16 thr/row, 4 elems
  const int scol = (tid & 15) * 4;

  for (int k0 = 0; k0 < HDIM; k0 += 64) {
    __syncthreads();
    if (k0 + 64 < HDIM)
      __builtin_prefetch(W + (size_t)(bn*128 + srow)*HDIM + k0 + 64 + scol, 0, 0);
    #pragma unroll
    for (int p = 0; p < 4; ++p) {
      const int r = arow + p*32;
      async_copy_b128(A + (size_t)(bm*128 + r)*HDIM + k0 + acol, &As[r][acol]);
    }
    #pragma unroll
    for (int p = 0; p < 8; ++p) {
      const int r = srow + p*16;
      vf4 fb = *(const vf4*)(W + (size_t)(bn*128 + r)*HDIM + k0 + scol);
      Pack4 pb;
      pb.h[0]=(__bf16)fb.x; pb.h[1]=(__bf16)fb.y; pb.h[2]=(__bf16)fb.z; pb.h[3]=(__bf16)fb.w;
      *(vu2*)&Bs[r][scol] = pb.u;
    }
    async_wait();
    __syncthreads();

    #pragma unroll
    for (int kk = 0; kk < 64; kk += 32) {
      Frag af[4], bf2[2];
      #pragma unroll
      for (int mt = 0; mt < 4; ++mt) {
        const __bf16* p = &As[wm*64 + mt*16 + ml][kk + half*8];
        af[mt].u[0] = *(const vu4*)p;
        af[mt].u[1] = *(const vu4*)(p + 16);
      }
      #pragma unroll
      for (int nt = 0; nt < 2; ++nt) {
        const __bf16* p = &Bs[wn*32 + nt*16 + ml][kk + half*8];
        bf2[nt].u[0] = *(const vu4*)p;
        bf2[nt].u[1] = *(const vu4*)(p + 16);
      }
      #pragma unroll
      for (int mt = 0; mt < 4; ++mt)
        #pragma unroll
        for (int nt = 0; nt < 2; ++nt)
          acc[mt*2+nt] = wmma_bf16(af[mt], bf2[nt], acc[mt*2+nt]);
    }
  }

  #pragma unroll
  for (int mt = 0; mt < 4; ++mt)
    #pragma unroll
    for (int nt = 0; nt < 2; ++nt)
      #pragma unroll
      for (int r = 0; r < 8; ++r) {
        const int m = bm*128 + wm*64 + mt*16 + r + half*8;
        const int n = bn*128 + wn*32 + nt*16 + ml;
        out[(size_t)m*HDIM + n] = acc[mt*2+nt][r];
      }
}

// =====================================================================
extern "C" void kernel_launch(void* const* d_in, const int* in_sizes, int n_in,
                              void* d_out, int out_size, void* d_ws, size_t ws_size,
                              hipStream_t stream) {
  const float* hs   = (const float*)d_in[0];  // (S,B,H)
  // d_in[1]: attention_mask (causal, known statically) -- unused
  const float* rope = (const float*)d_in[2];  // (S,1,32,2)
  const float* Wq   = (const float*)d_in[3];  // (4608,4096)
  const float* bq   = (const float*)d_in[4];  // (4608,)
  const float* Wd   = (const float*)d_in[5];  // (4096,4096)
  float* out = (float*)d_out;

  char* ws = (char*)d_ws;
  const size_t SZ_Q = (size_t)BATCH*NHEADS *S_LEN*HEADD*2;  // 32 MiB
  const size_t SZ_K = (size_t)BATCH*NGROUPS*S_LEN*HEADD*2;  //  2 MiB
  __bf16* Qb = (__bf16*)(ws);
  __bf16* Kb = (__bf16*)(ws + SZ_Q);
  __bf16* Vb = (__bf16*)(ws + SZ_Q + SZ_K);
  __bf16* Cx = (__bf16*)(ws + SZ_Q + 2*SZ_K);               // 32 MiB more

  qkv_gemm_kernel<<<dim3(NQKV/128, MROWS/128), 256, 0, stream>>>(hs, Wq, bq, Qb, Kb, Vb);

  const int npairs = BATCH*NHEADS*S_LEN*32 + BATCH*NGROUPS*S_LEN*32;
  rope_kernel<<<(npairs + 255)/256, 256, 0, stream>>>(Qb, Kb, rope);

  attn_kernel<<<dim3(S_LEN/64, BATCH*NHEADS), 128, 0, stream>>>(Qb, Kb, Vb, Cx);

  dense_gemm_kernel<<<dim3(HDIM/128, MROWS/128), 256, 0, stream>>>(Cx, Wd, out);
}
